// SteerableDecoder_80066780332740
// MI455X (gfx1250) — matrix operand order repaired and checked by reading, not compiled, not run on hardware
//
#include <hip/hip_runtime.h>
#include <hip/hip_bf16.h>
#include <cmath>
#include <cstdint>

// ---------------------------------------------------------------------------
// Steerable GNN decoder for MI455X (gfx1250, wave32).
// Channel-mixing GEMMs of all tensor products -> v_wmma_f32_16x16x32_bf16.
// Weights pre-converted to bf16 [V][Ustride]; B tiles staged to LDS with
// global_load_async_to_lds_b128 (ASYNCcnt), double buffered so the copy
// overlaps the WMMAs and the next chunk's A-fragment gathers.
// ---------------------------------------------------------------------------

typedef __attribute__((ext_vector_type(16))) __bf16 v16bf_t;
typedef __attribute__((ext_vector_type(8)))  float  v8f;

#define USE_ASYNC_STAGE 1

// ----- problem sizes (fixed by the reference) -----
#define GRID_N   1728      // 12^3
#define CODE_DIM 128
#define NB       2
#define NQ       2048
#define KNN      32
#define NQTOT    (NB*NQ)              // 4096
#define NTOT     (NQTOT + NB*GRID_N)  // 7552
#define NEDGE    (NQTOT*KNN)          // 131072
#define EC       8192                 // edges per chunk
#define NCHUNK   (NEDGE/EC)           // 16
#define H0_      128
#define H1_      64
#define H2_      32
#define NGATES_  96
#define CUTOFF_  8.0f
#define EPS_     1e-5f

#define C0_  0.28209479177387814f
#define C1_  0.4886025119029199f
#define C2A_ 1.0925484305920792f
#define C2B_ 0.31539156525252005f
#define C2C_ 0.5462742152960396f

__device__ __forceinline__ unsigned short f2bf(float f) {
  unsigned u = __float_as_uint(f);
  unsigned r = u + 0x7FFFu + ((u >> 16) & 1u);
  return (unsigned short)(r >> 16);
}
__device__ __forceinline__ float sigm(float x) { return 1.f / (1.f + __expf(-x)); }

// ---------------------------------------------------------------------------
// Tensor-product descriptor (runtime-driven; kernel templated on L3, #V-tiles)
// ---------------------------------------------------------------------------
struct Seg   { const float* ptr; int U; int rowStride; int idxMode; }; // 0 direct,1 dst(edge>>5),2 srcIdx
struct PathD { Seg seg[3]; const unsigned short* W; int Ustride; int nSeg; int Utot; int l1; int l2; int tri; };
struct TpDesc {
  int N, V, nPath;
  float scale;
  PathD path[4];
  const int* srcIdx; int edgeBase;
  float attr0c;
  const float* attr1; const float* attr2;
  const float* w3j;          // [11][5][5][5]
  float* out; int outStride; // out[row*outStride + v*(2l3+1) + k]
  const float* bias;         // l3==0 only
};

template<int L3, int NVT>
__global__ __launch_bounds__(32) void k_tp(TpDesc d) {
  constexpr int KD = 2*L3 + 1;
  __shared__ float sM[16*5*5];
  __shared__ __align__(16) unsigned short sB[2][NVT*16*32];
  const int lane = threadIdx.x;
  const int m    = lane & 15;
  const int ph   = lane >> 4;
  const int rowBase = blockIdx.x * 16;
  const int row  = rowBase + m;
  const bool rowOk = row < d.N;

  // zero-pad LDS once (rows v >= V are never touched by the async copies)
  for (int t = lane; t < 2*NVT*16*32; t += 32) ((unsigned short*)sB)[t] = 0;
  __syncthreads();

  v8f acc[NVT][KD];
  const v8f vz = {0.f,0.f,0.f,0.f,0.f,0.f,0.f,0.f};
  #pragma unroll
  for (int vt = 0; vt < NVT; ++vt)
    #pragma unroll
    for (int k = 0; k < KD; ++k) acc[vt][k] = vz;

  for (int p = 0; p < d.nPath; ++p) {
    const PathD P = d.path[p];
    const int ND1 = 2*P.l1 + 1;
    const int ND2 = 2*P.l2 + 1;

    // warm L2 with this path's weights (global_prefetch_b8)
    __builtin_prefetch(P.W + (size_t)lane * 64, 0, 1);

    // ---- per-row mix matrix  mmat[r][i][k] = sum_j attr[r][j] * W3J[i][j][k]
    __syncthreads();
    for (int t = lane; t < 16*ND1*KD; t += 32) {
      int r = t / (ND1*KD); int rem = t - r*(ND1*KD);
      int i = rem / KD;     int k   = rem - i*KD;
      float a = 0.f;
      int gr = rowBase + r;
      if (gr < d.N) {
        const float* C = d.w3j + P.tri*125 + i*25 + k;
        for (int j = 0; j < ND2; ++j) {
          float av = (P.l2 == 0) ? d.attr0c
                   : (P.l2 == 1) ? d.attr1[gr*3 + j]
                                 : d.attr2[gr*5 + j];
          a += av * C[j*5];
        }
      }
      sM[t] = a;
    }
    __syncthreads();

    // stage one 32-deep K-chunk of B into LDS buffer `buf` (async, per-lane rows)
    auto stageB = [&](int u0, int buf) {
#if USE_ASYNC_STAGE
      for (int v = lane; v < d.V; v += 32) {
        unsigned ldsa = (unsigned)(unsigned long long)&sB[buf][v*32];
        unsigned long long ga =
            (unsigned long long)(P.W + (size_t)v * P.Ustride + u0);
        asm volatile(
          "global_load_async_to_lds_b128 %0, %1, off\n\t"
          "global_load_async_to_lds_b128 %0, %1, off offset:16\n\t"
          "global_load_async_to_lds_b128 %0, %1, off offset:32\n\t"
          "global_load_async_to_lds_b128 %0, %1, off offset:48"
          :: "v"(ldsa), "v"(ga) : "memory");
      }
#else
      for (int t = lane; t < NVT*16*32; t += 32) {
        int uo = t & 31; int v = t >> 5;
        sB[buf][t] = (v < d.V) ? P.W[(size_t)v * P.Ustride + u0 + uo]
                               : (unsigned short)0;
      }
#endif
    };

    int buf = 0;
    stageB(0, 0);

    for (int u0 = 0; u0 < P.Utot; u0 += 32) {
      // ---- build A fragments (16-bit A 16x32 layout, doc 7.12.2);
      //      the scattered gathers overlap the async B copy in flight
      union AFrag { unsigned u[8]; v16bf_t v; } af[KD];
      #pragma unroll
      for (int k = 0; k < KD; ++k)
        #pragma unroll
        for (int q = 0; q < 8; ++q) af[k].u[q] = 0u;

      #pragma unroll
      for (int elem = 0; elem < 16; ++elem) {
        const int preg = elem >> 1, eo = elem & 1;
        const int kpos = ((preg >> 2) << 4) | (ph << 3) | ((preg & 3) << 1) | eo;
        int u = u0 + kpos;
        if (rowOk && u < P.Utot) {
          int ul = u; int si = 0;
          while (si < P.nSeg - 1 && ul >= P.seg[si].U) { ul -= P.seg[si].U; ++si; }
          const Seg S = P.seg[si];
          long idx;
          if (S.idxMode == 0)      idx = row;
          else if (S.idxMode == 1) idx = (long)((d.edgeBase + row) >> 5);
          else                     idx = (long)d.srcIdx[d.edgeBase + row];
          const float* xp = S.ptr + idx * (long)S.rowStride + ul * ND1;
          float xv[5];
          for (int i = 0; i < ND1; ++i) xv[i] = xp[i];
          #pragma unroll
          for (int k = 0; k < KD; ++k) {
            float tv = 0.f;
            for (int i = 0; i < ND1; ++i) tv += xv[i] * sM[(m*ND1 + i)*KD + k];
            af[k].u[preg] |= ((unsigned)f2bf(tv)) << (eo * 16);
          }
        }
      }

#if USE_ASYNC_STAGE
      asm volatile("s_wait_asynccnt 0" ::: "memory");
#endif
      __syncthreads();

      // kick off the next chunk's copy into the other buffer; it overlaps
      // this chunk's WMMAs and the next chunk's A gathers
      if (u0 + 32 < P.Utot) stageB(u0 + 32, buf ^ 1);

      // ---- WMMA over all V tiles (B fragment reused across the KD GEMMs)
      #pragma unroll
      for (int vt = 0; vt < NVT; ++vt) {
        union BFrag { unsigned u[8]; v16bf_t v; uint4 q[2]; } bfr;
        const unsigned short* bp = &sB[buf][(vt*16 + (lane & 15))*32 + ph*8];
        bfr.q[0] = *(const uint4*)bp;
        bfr.q[1] = *(const uint4*)(bp + 16);
        #pragma unroll
        for (int k = 0; k < KD; ++k)
          acc[vt][k] = __builtin_amdgcn_wmma_f32_16x16x32_bf16(
              false, af[k].v, false, bfr.v, (short)0, acc[vt][k], false, false);
      }
      buf ^= 1;
      __syncthreads();
    }
  }

  // ---- epilogue: f32 C/D layout (doc 7.12.2) -> out[row][v][k]
  const int n = lane & 15;
  #pragma unroll
  for (int vt = 0; vt < NVT; ++vt) {
    int v = vt*16 + n;
    if (v >= d.V) continue;
    #pragma unroll
    for (int r = 0; r < 8; ++r) {
      int orow = rowBase + r + ph*8;
      if (orow >= d.N) continue;
      float* op = d.out + (long)orow * d.outStride + v*KD;
      #pragma unroll
      for (int k = 0; k < KD; ++k) {
        float val = acc[vt][k][r] * d.scale;
        if (L3 == 0 && d.bias) val += d.bias[v];
        op[k] = val;
      }
    }
  }
}

// weight pre-conversion: f32 [U][V] -> bf16 [V][Ustride] (zero-padded tail)
__global__ void k_wcvt(const float* W, unsigned short* o, int U, int V, int Ust) {
  long idx = (long)blockIdx.x*256 + threadIdx.x;
  if (idx >= (long)V*Ust) return;
  int v = (int)(idx / Ust), u = (int)(idx % Ust);
  o[idx] = (u < U) ? f2bf(W[(long)u*V + v]) : (unsigned short)0;
}

// ---------------------------------------------------------------------------
// W3J table (analytic, reference SH ordering: l1=(x,y,z), l2=(xy,yz,3z^2-1,xz,x^2-y^2))
// ---------------------------------------------------------------------------
__global__ void k_w3j(float* w) {
  if (threadIdx.x != 0 || blockIdx.x != 0) return;
  for (int i = 0; i < 11*125; ++i) w[i] = 0.f;
  auto S = [&](int t,int i,int j,int k,float v){ w[t*125 + i*25 + j*5 + k] = v; };
  const float r3 = 0.57735026919f, r5 = 0.44721359549f;
  S(0,0,0,0,1.f);
  for (int j = 0; j < 3; ++j) S(1,0,j,j,r3);
  for (int j = 0; j < 5; ++j) S(2,0,j,j,r5);
  for (int i = 0; i < 3; ++i) S(3,i,0,i,r3);
  for (int i = 0; i < 3; ++i) S(4,i,i,0,r3);
  const float al = 0.31622776601f, be = 0.18257418583f;      // (1,1,2)
  S(5,0,1,0,al); S(5,1,0,0,al);
  S(5,1,2,1,al); S(5,2,1,1,al);
  S(5,0,0,2,-be); S(5,1,1,2,-be); S(5,2,2,2,2.f*be);
  S(5,0,2,3,al); S(5,2,0,3,al);
  S(5,0,0,4,al); S(5,1,1,4,-al);
  for (int i=0;i<3;++i) for(int j=0;j<3;++j) for(int k=0;k<5;++k) {
    float v = w[5*125 + i*25 + j*5 + k];
    if (v != 0.f) {
      w[6*125 + i*25 + k*5 + j] = v;   // (1,2,1)
      w[8*125 + k*25 + i*5 + j] = v;   // (2,1,1)
    }
  }
  for (int i = 0; i < 5; ++i) S(7,i,0,i,r5);
  for (int i = 0; i < 5; ++i) S(9,i,i,0,r5);
  const float s22 = 0.20701966780f, a22 = 0.23904572186f, b22 = 0.11952286093f; // (2,2,2)
  auto P3 = [&](int a,int b,int c,float v){
    int id[3] = {a,b,c};
    const int pm[6][3] = {{0,1,2},{0,2,1},{1,0,2},{1,2,0},{2,0,1},{2,1,0}};
    for (int p = 0; p < 6; ++p)
      w[10*125 + id[pm[p][0]]*25 + id[pm[p][1]]*5 + id[pm[p][2]]] = v;
  };
  P3(2,2,2,a22); P3(0,0,2,-a22); P3(1,1,2,b22); P3(3,3,2,b22); P3(4,4,2,-a22);
  P3(0,1,3,s22); P3(1,1,4,-s22); P3(3,3,4,s22);
}

// ---------------------------------------------------------------------------
// kNN + edge attrs (grid staged in LDS, per-thread insertion top-32)
// ---------------------------------------------------------------------------
__global__ __launch_bounds__(256) void k_knn(const float* qp, const float* grid,
    int* srcIdx, float* mask, float* d2e, float* sh1, float* sh2) {
  __shared__ float sg[GRID_N*3];
  for (int t = threadIdx.x; t < GRID_N*3; t += 256) sg[t] = grid[t];
  __syncthreads();
  int q = blockIdx.x*256 + threadIdx.x;
  if (q >= NQTOT) return;
  float qx = qp[q*3], qy = qp[q*3+1], qz = qp[q*3+2];
  float bd[KNN]; int bi[KNN];
  for (int k = 0; k < KNN; ++k) { bd[k] = 3.4e38f; bi[k] = 0; }
  for (int g = 0; g < GRID_N; ++g) {
    float dx = sg[g*3]-qx, dy = sg[g*3+1]-qy, dz = sg[g*3+2]-qz;
    float d2 = dx*dx + dy*dy + dz*dz;
    if (d2 < bd[KNN-1]) {
      int pos = KNN-1;
      while (pos > 0 && bd[pos-1] > d2) { bd[pos]=bd[pos-1]; bi[pos]=bi[pos-1]; --pos; }
      bd[pos] = d2; bi[pos] = g;
    }
  }
  int b = q / NQ;
  for (int k = 0; k < KNN; ++k) {
    int e = q*KNN + k, g = bi[k];
    float rx = sg[g*3]-qx, ry = sg[g*3+1]-qy, rz = sg[g*3+2]-qz;
    float d2 = bd[k], dist = sqrtf(d2);
    mask[e] = (dist <= CUTOFF_) ? 1.f : 0.f;
    d2e[e]  = d2;
    srcIdx[e] = NQTOT + b*GRID_N + g;
    float inv = 1.f / fmaxf(dist, 1e-12f);
    float ux = rx*inv, uy = ry*inv, uz = rz*inv;
    sh1[e*3+0] = C1_*ux; sh1[e*3+1] = C1_*uy; sh1[e*3+2] = C1_*uz;
    sh2[e*5+0] = C2A_*ux*uy; sh2[e*5+1] = C2A_*uy*uz;
    sh2[e*5+2] = C2B_*(3.f*uz*uz - 1.f);
    sh2[e*5+3] = C2A_*ux*uz; sh2[e*5+4] = C2C_*(ux*ux - uy*uy);
  }
}

__global__ void k_node_attr(const float* mask, const float* sh1, const float* sh2,
                            float* na1, float* na2) {
  int n = blockIdx.x*256 + threadIdx.x;
  if (n >= NTOT) return;
  if (n < NQTOT) {
    float cnt = 0.f, a1[3] = {0,0,0}, a2[5] = {0,0,0,0,0};
    for (int k = 0; k < KNN; ++k) {
      int e = n*KNN + k; float mk = mask[e]; cnt += mk;
      for (int j = 0; j < 3; ++j) a1[j] += mk*sh1[e*3+j];
      for (int j = 0; j < 5; ++j) a2[j] += mk*sh2[e*5+j];
    }
    float ic = 1.f / fmaxf(cnt, 1.f);
    for (int j = 0; j < 3; ++j) na1[n*3+j] = a1[j]*ic;
    for (int j = 0; j < 5; ++j) na2[n*5+j] = a2[j]*ic;
  } else {
    for (int j = 0; j < 3; ++j) na1[n*3+j] = 0.f;
    for (int j = 0; j < 5; ++j) na2[n*5+j] = 0.f;
  }
}

__global__ void k_fill_x0in(const float* codes, float* x0in) {
  long idx = (long)blockIdx.x*256 + threadIdx.x;
  if (idx >= (long)NTOT*CODE_DIM) return;
  int n = (int)(idx / CODE_DIM), c = (int)(idx % CODE_DIM);
  x0in[idx] = (n < NQTOT) ? 0.f : codes[(long)(n - NQTOT)*CODE_DIM + c];
}

__global__ void k_zero(float* p, long n) {
  long i = (long)blockIdx.x*256 + threadIdx.x;
  if (i < n) p[i] = 0.f;
}

// gate: silu scalars in place, sigmoid-gate l=1/2 features
__global__ void k_gate(int N, float* g0, float* g1, float* g2) {
  long idx = (long)blockIdx.x*256 + threadIdx.x;
  if (idx >= (long)N*224) return;
  int r = (int)(idx / 224), c = (int)(idx % 224);
  if (c < 128) {
    float v = g0[(long)r*224 + c];
    g0[(long)r*224 + c] = v * sigm(v);
  } else if (c < 192) {
    int v = c - 128;
    float g = sigm(g0[(long)r*224 + c]);
    for (int i = 0; i < 3; ++i) g1[(long)r*192 + v*3 + i] *= g;
  } else {
    int v = c - 192;
    float g = sigm(g0[(long)r*224 + c]);
    for (int i = 0; i < 5; ++i) g2[(long)r*160 + v*5 + i] *= g;
  }
}

// aggregate masked messages of one chunk into query nodes (no atomics: edges
// of query n are rows n*32..n*32+31)
__global__ void k_aggregate(int ch0, const float* mask,
                            const float* h0, const float* h1, const float* h2,
                            float* a0, float* a1, float* a2) {
  long idx = (long)blockIdx.x*256 + threadIdx.x;
  if (idx >= (long)(EC/KNN)*480) return;
  int qi = (int)(idx / 480), c = (int)(idx % 480);
  int n = ch0/KNN + qi;
  float s = 0.f;
  if (c < 128) {
    for (int k = 0; k < KNN; ++k) { long r = qi*KNN + k; s += mask[ch0+r]*h0[r*224 + c]; }
    a0[(long)n*128 + c] = s;
  } else if (c < 320) {
    int i = c - 128;
    for (int k = 0; k < KNN; ++k) { long r = qi*KNN + k; s += mask[ch0+r]*h1[r*192 + i]; }
    a1[(long)n*192 + i] = s;
  } else {
    int i = c - 320;
    for (int k = 0; k < KNN; ++k) { long r = qi*KNN + k; s += mask[ch0+r]*h2[r*160 + i]; }
    a2[(long)n*160 + i] = s;
  }
}

__global__ void k_residual(float* x0, float* x1, float* x2,
                           const float* u0, const float* u1, const float* u2) {
  long idx = (long)blockIdx.x*256 + threadIdx.x;
  if (idx >= (long)NTOT*480) return;
  int n = (int)(idx / 480), c = (int)(idx % 480);
  if (c < 128)      { long o = (long)n*128 + c;        x0[o] += u0[o]; }
  else if (c < 320) { long o = (long)n*192 + (c-128);  x1[o] += u1[o]; }
  else              { long o = (long)n*160 + (c-320);  x2[o] += u2[o]; }
}

__global__ __launch_bounds__(256) void k_stats(const float* x0, const float* x1,
                                               const float* x2, float* stats) {
  __shared__ float s1[256], s2[256];
  int c = blockIdx.x, b = blockIdx.y, tid = threadIdx.x;
  const int NPB = NQ + GRID_N; // 3776 nodes per batch
  float a1 = 0.f, a2 = 0.f;
  for (int t = tid; t < NPB; t += 256) {
    int n = (t < NQ) ? b*NQ + t : NQTOT + b*GRID_N + (t - NQ);
    if (c < 128)      { float v = x0[(long)n*128 + c]; a1 += v; a2 += v*v; }
    else if (c < 192) { int v = c-128; float s = 0.f;
                        for (int i = 0; i < 3; ++i) { float w = x1[(long)n*192 + v*3 + i]; s += w*w; }
                        a1 += s*(1.f/3.f); }
    else              { int v = c-192; float s = 0.f;
                        for (int i = 0; i < 5; ++i) { float w = x2[(long)n*160 + v*5 + i]; s += w*w; }
                        a1 += s*0.2f; }
  }
  s1[tid] = a1; s2[tid] = a2; __syncthreads();
  for (int o = 128; o > 0; o >>= 1) {
    if (tid < o) { s1[tid] += s1[tid+o]; s2[tid] += s2[tid+o]; }
    __syncthreads();
  }
  if (tid == 0) {
    float inv = 1.f / (float)NPB;
    if (c < 128) {
      float mu = s1[0]*inv, var = s2[0]*inv - mu*mu;
      stats[(b*224+c)*2] = mu; stats[(b*224+c)*2+1] = rsqrtf(var + EPS_);
    } else {
      stats[(b*224+c)*2] = 0.f; stats[(b*224+c)*2+1] = rsqrtf(s1[0]*inv + EPS_);
    }
  }
}

__global__ void k_instnorm(float* x0, float* x1, float* x2, const float* stats) {
  long idx = (long)blockIdx.x*256 + threadIdx.x;
  if (idx >= (long)NTOT*224) return;
  int n = (int)(idx / 224), c = (int)(idx % 224);
  int b = (n < NQTOT) ? n / NQ : (n - NQTOT) / GRID_N;
  float mu = stats[(b*224+c)*2], rs = stats[(b*224+c)*2+1];
  if (c < 128)      x0[(long)n*128 + c] = (x0[(long)n*128 + c] - mu)*rs;
  else if (c < 192) { int v = c-128; for (int i = 0; i < 3; ++i) x1[(long)n*192 + v*3 + i] *= rs; }
  else              { int v = c-192; for (int i = 0; i < 5; ++i) x2[(long)n*160 + v*5 + i] *= rs; }
}

__global__ void k_final(const float* postOut, float* out) {
  long i = (long)blockIdx.x*256 + threadIdx.x;
  if (i < (long)NQTOT*24) out[i] = postOut[i];
}

// ---------------------------------------------------------------------------
// Host-side wiring
// ---------------------------------------------------------------------------
struct TpW {
  const float* b; int nw;
  const float* W[11];
  const unsigned short* Wbf[11];
  int Ust[11], U[11], V[11];
  int l1[11], l2[11], l3[11];
};

static int iabs_(int a) { return a < 0 ? -a : a; }

static void take_tp(void* const* d_in, int& cur, TpW& t,
                    const int* inMul, const int* outMul, bool bias) {
  t.b = bias ? (const float*)d_in[cur++] : nullptr;
  t.nw = 0;
  for (int a = 0; a < 3; ++a)
    for (int b2 = 0; b2 < 3; ++b2)
      for (int c = 0; c < 3; ++c)
        if (iabs_(a-b2) <= c && c <= a+b2 && ((a+b2+c)&1) == 0 &&
            inMul[a] > 0 && outMul[c] > 0) {
          t.W[t.nw] = (const float*)d_in[cur++];
          t.l1[t.nw] = a; t.l2[t.nw] = b2; t.l3[t.nw] = c;
          t.U[t.nw] = inMul[a]; t.V[t.nw] = outMul[c];
          ++t.nw;
        }
}

static int tri_index(int a, int b, int c) {
  int t = 0;
  for (int i = 0; i < 3; ++i)
    for (int j = 0; j < 3; ++j)
      for (int k = 0; k < 3; ++k)
        if (iabs_(i-j) <= k && k <= i+j && ((i+j+k)&1) == 0) {
          if (i == a && j == b && k == c) return t;
          ++t;
        }
  return 0;
}

static inline int g256(long n) { return (int)((n + 255) / 256); }

static void cvt_weights(TpW& w, unsigned short*& base, hipStream_t stream) {
  for (int i = 0; i < w.nw; ++i) {
    int U = w.U[i], V = w.V[i];
    int Ust = (U + 31) & ~31;
    w.Ust[i] = Ust; w.Wbf[i] = base;
    k_wcvt<<<g256((long)V*Ust), 256, 0, stream>>>(w.W[i], base, U, V, Ust);
    base += (size_t)V * Ust;
  }
}

struct TpIn {
  Seg segs[3][3]; int nSeg[3];
  float attr0c; const float* attr1; const float* attr2;
  const int* srcIdx; int edgeBase;
  int N;
};

static void launch_tp(const TpIn& in, const TpW& w, const int* outMul,
                      float* const out[3], const int outStride[3],
                      const float* w3j, hipStream_t stream) {
  for (int l3 = 0; l3 < 3; ++l3) {
    if (outMul[l3] == 0) continue;
    TpDesc d{};
    d.N = in.N; d.V = outMul[l3]; d.nPath = 0;
    int fan = 0;
    for (int i = 0; i < w.nw; ++i) {
      if (w.l3[i] != l3) continue;
      PathD& P = d.path[d.nPath++];
      int l1 = w.l1[i];
      P.nSeg = in.nSeg[l1]; P.l1 = l1; P.l2 = w.l2[i];
      P.tri = tri_index(l1, w.l2[i], l3);
      P.W = w.Wbf[i]; P.Ustride = w.Ust[i];
      int ut = 0;
      for (int s = 0; s < P.nSeg; ++s) { P.seg[s] = in.segs[l1][s]; ut += P.seg[s].U; }
      P.Utot = ut; fan += ut;
    }
    if (d.nPath == 0) continue;
    d.scale = 1.0f / sqrtf((float)fan);
    d.srcIdx = in.srcIdx; d.edgeBase = in.edgeBase;
    d.attr0c = in.attr0c; d.attr1 = in.attr1; d.attr2 = in.attr2;
    d.w3j = w3j; d.out = out[l3]; d.outStride = outStride[l3];
    d.bias = (l3 == 0) ? w.b : nullptr;
    int grid = (d.N + 15) / 16;
    int nvt = (d.V + 15) / 16;
    if (l3 == 0 && nvt == 14)     k_tp<0,14><<<grid, 32, 0, stream>>>(d);
    else if (l3 == 0 && nvt == 8) k_tp<0, 8><<<grid, 32, 0, stream>>>(d);
    else if (l3 == 1 && nvt == 4) k_tp<1, 4><<<grid, 32, 0, stream>>>(d);
    else if (l3 == 1 && nvt == 1) k_tp<1, 1><<<grid, 32, 0, stream>>>(d);
    else if (l3 == 2 && nvt == 2) k_tp<2, 2><<<grid, 32, 0, stream>>>(d);
  }
}

extern "C" void kernel_launch(void* const* d_in, const int* in_sizes, int n_in,
                              void* d_out, int out_size, void* d_ws, size_t ws_size,
                              hipStream_t stream) {
  const float* qp    = (const float*)d_in[0];
  const float* codes = (const float*)d_in[1];
  const float* gridc = (const float*)d_in[2];
  int cur = 3;

  const int mulHID[3]   = {H0_, H1_, H2_};
  const int mulGate[3]  = {H0_ + NGATES_, H1_, H2_};
  const int mulCode[3]  = {CODE_DIM, 0, 0};
  const int mulM1[3]    = {2*H0_ + 1, 2*H1_, 2*H2_};
  const int mulU1[3]    = {2*H0_, 2*H1_, 2*H2_};
  const int mulPost[3]  = {0, 8, 0};

  TpW wEmbed; take_tp(d_in, cur, wEmbed, mulCode, mulHID, true);
  TpW wM1[5], wM2[5], wU1[5], wU2[5];
  for (int L = 0; L < 5; ++L) {
    take_tp(d_in, cur, wM1[L], mulM1,  mulGate, true);
    take_tp(d_in, cur, wM2[L], mulHID, mulGate, true);
    take_tp(d_in, cur, wU1[L], mulU1,  mulGate, true);
    take_tp(d_in, cur, wU2[L], mulHID, mulHID,  true);
  }
  TpW wPost; take_tp(d_in, cur, wPost, mulHID, mulPost, false);
  TpW wPre;  take_tp(d_in, cur, wPre,  mulHID, mulGate, true);
  (void)n_in; (void)in_sizes; (void)out_size;

  // ---- workspace layout (floats, 16B-aligned blocks) ----
  float* Wf = (float*)d_ws;
  size_t o = 0;
  auto alloc = [&](size_t n) { float* p = Wf + o; o += (n + 3) & ~(size_t)3; return p; };
  float* w3j  = alloc(11*125);
  int*   srcI = (int*)alloc(NEDGE);
  float* mask = alloc(NEDGE);
  float* d2e  = alloc(NEDGE);
  float* sh1  = alloc((size_t)NEDGE*3);
  float* sh2  = alloc((size_t)NEDGE*5);
  float* na1  = alloc((size_t)NTOT*3);
  float* na2  = alloc((size_t)NTOT*5);
  float* x0   = alloc((size_t)NTOT*128);
  float* x1   = alloc((size_t)NTOT*192);
  float* x2   = alloc((size_t)NTOT*160);
  float* agg0 = alloc((size_t)NTOT*128);
  float* agg1 = alloc((size_t)NTOT*192);
  float* agg2 = alloc((size_t)NTOT*160);
  float* nG0  = alloc((size_t)NTOT*224);
  float* nG1  = alloc((size_t)NTOT*192);
  float* nG2  = alloc((size_t)NTOT*160);
  float* upd0 = alloc((size_t)NTOT*128);
  float* upd1 = alloc((size_t)NTOT*192);
  float* upd2 = alloc((size_t)NTOT*160);
  float* x0in = alloc((size_t)NTOT*128);
  float* stats= alloc(2*224*2);
  float* postO= alloc((size_t)NTOT*24);
  float* eG0  = alloc((size_t)EC*224);
  float* eG1  = alloc((size_t)EC*192);
  float* eG2  = alloc((size_t)EC*160);
  float* eH0  = alloc((size_t)EC*224);
  float* eH1  = alloc((size_t)EC*192);
  float* eH2  = alloc((size_t)EC*160);
  unsigned short* wbfBase = (unsigned short*)alloc(1600000); // 3.2M bf16 >= 2.36M needed
  if (o * sizeof(float) > ws_size) return;  // insufficient scratch

  // ---- one-time weight conversion to bf16 [V][Ustride] ----
  {
    unsigned short* base = wbfBase;
    cvt_weights(wEmbed, base, stream);
    for (int L = 0; L < 5; ++L) {
      cvt_weights(wM1[L], base, stream);
      cvt_weights(wM2[L], base, stream);
      cvt_weights(wU1[L], base, stream);
      cvt_weights(wU2[L], base, stream);
    }
    cvt_weights(wPost, base, stream);
    cvt_weights(wPre,  base, stream);
  }

  // ---- preprocessing ----
  k_w3j<<<1, 1, 0, stream>>>(w3j);
  k_knn<<<NQTOT/256, 256, 0, stream>>>(qp, gridc, srcI, mask, d2e, sh1, sh2);
  k_node_attr<<<g256(NTOT), 256, 0, stream>>>(mask, sh1, sh2, na1, na2);
  k_fill_x0in<<<g256((long)NTOT*128), 256, 0, stream>>>(codes, x0in);

  // ---- embed TP: codes -> x ----
  TpIn cfgN{};  // node-indexed, node attrs
  cfgN.attr0c = 1.f; cfgN.attr1 = na1; cfgN.attr2 = na2;
  cfgN.srcIdx = nullptr; cfgN.edgeBase = 0; cfgN.N = NTOT;
  {
    TpIn c = cfgN;
    c.nSeg[0] = 1; c.segs[0][0] = {x0in, 128, 128, 0};
    c.nSeg[1] = 0; c.nSeg[2] = 0;
    float* outs[3] = {x0, x1, x2}; const int ostr[3] = {128, 192, 160};
    launch_tp(c, wEmbed, mulHID, outs, ostr, w3j, stream);
  }

  for (int L = 0; L < 5; ++L) {
    k_zero<<<g256((long)NTOT*480), 256, 0, stream>>>(agg0, (long)NTOT*480); // agg0/1/2 contiguous
    for (int ch = 0; ch < NCHUNK; ++ch) {
      int ch0 = ch * EC;
      // --- m1: edge TP on (xi || xj || d2) ---
      TpIn c{};
      c.attr0c = C0_; c.attr1 = sh1 + (size_t)ch0*3; c.attr2 = sh2 + (size_t)ch0*5;
      c.srcIdx = srcI; c.edgeBase = ch0; c.N = EC;
      c.nSeg[0] = 3;
      c.segs[0][0] = {x0, 128, 128, 1}; c.segs[0][1] = {x0, 128, 128, 2};
      c.segs[0][2] = {d2e + ch0, 1, 1, 0};
      c.nSeg[1] = 2; c.segs[1][0] = {x1, 64, 192, 1}; c.segs[1][1] = {x1, 64, 192, 2};
      c.nSeg[2] = 2; c.segs[2][0] = {x2, 32, 160, 1}; c.segs[2][1] = {x2, 32, 160, 2};
      float* og[3] = {eG0, eG1, eG2}; const int osg[3] = {224, 192, 160};
      launch_tp(c, wM1[L], mulGate, og, osg, w3j, stream);
      k_gate<<<g256((long)EC*224), 256, 0, stream>>>(EC, eG0, eG1, eG2);
      // --- m2: edge TP on gated m1 output ---
      TpIn c2{};
      c2.attr0c = C0_; c2.attr1 = c.attr1; c2.attr2 = c.attr2;
      c2.srcIdx = srcI; c2.edgeBase = ch0; c2.N = EC;
      c2.nSeg[0] = 1; c2.segs[0][0] = {eG0, 128, 224, 0};
      c2.nSeg[1] = 1; c2.segs[1][0] = {eG1, 64, 192, 0};
      c2.nSeg[2] = 1; c2.segs[2][0] = {eG2, 32, 160, 0};
      float* oh[3] = {eH0, eH1, eH2};
      launch_tp(c2, wM2[L], mulGate, oh, osg, w3j, stream);
      k_gate<<<g256((long)EC*224), 256, 0, stream>>>(EC, eH0, eH1, eH2);
      k_aggregate<<<g256((long)(EC/KNN)*480), 256, 0, stream>>>(
          ch0, mask, eH0, eH1, eH2, agg0, agg1, agg2);
    }
    // --- u1 on (x || agg), u2, residual, instance norm ---
    {
      TpIn c = cfgN;
      c.nSeg[0] = 2; c.segs[0][0] = {x0, 128, 128, 0}; c.segs[0][1] = {agg0, 128, 128, 0};
      c.nSeg[1] = 2; c.segs[1][0] = {x1, 64, 192, 0};  c.segs[1][1] = {agg1, 64, 192, 0};
      c.nSeg[2] = 2; c.segs[2][0] = {x2, 32, 160, 0};  c.segs[2][1] = {agg2, 32, 160, 0};
      float* og[3] = {nG0, nG1, nG2}; const int osg[3] = {224, 192, 160};
      launch_tp(c, wU1[L], mulGate, og, osg, w3j, stream);
    }
    k_gate<<<g256((long)NTOT*224), 256, 0, stream>>>(NTOT, nG0, nG1, nG2);
    {
      TpIn c = cfgN;
      c.nSeg[0] = 1; c.segs[0][0] = {nG0, 128, 224, 0};
      c.nSeg[1] = 1; c.segs[1][0] = {nG1, 64, 192, 0};
      c.nSeg[2] = 1; c.segs[2][0] = {nG2, 32, 160, 0};
      float* ou[3] = {upd0, upd1, upd2}; const int osu[3] = {128, 192, 160};
      launch_tp(c, wU2[L], mulHID, ou, osu, w3j, stream);
    }
    k_residual<<<g256((long)NTOT*480), 256, 0, stream>>>(x0, x1, x2, upd0, upd1, upd2);
    k_stats<<<dim3(224, 2), 256, 0, stream>>>(x0, x1, x2, stats);
    k_instnorm<<<g256((long)NTOT*224), 256, 0, stream>>>(x0, x1, x2, stats);
  }

  // ---- pre (gated) + post -> [NTOT][8][3] ----
  {
    TpIn c = cfgN;
    c.nSeg[0] = 1; c.segs[0][0] = {x0, 128, 128, 0};
    c.nSeg[1] = 1; c.segs[1][0] = {x1, 64, 192, 0};
    c.nSeg[2] = 1; c.segs[2][0] = {x2, 32, 160, 0};
    float* og[3] = {nG0, nG1, nG2}; const int osg[3] = {224, 192, 160};
    launch_tp(c, wPre, mulGate, og, osg, w3j, stream);
  }
  k_gate<<<g256((long)NTOT*224), 256, 0, stream>>>(NTOT, nG0, nG1, nG2);
  {
    TpIn c = cfgN;
    c.nSeg[0] = 1; c.segs[0][0] = {nG0, 128, 224, 0};
    c.nSeg[1] = 1; c.segs[1][0] = {nG1, 64, 192, 0};
    c.nSeg[2] = 1; c.segs[2][0] = {nG2, 32, 160, 0};
    float* op[3] = {nullptr, postO, nullptr}; const int osp[3] = {0, 24, 0};
    launch_tp(c, wPost, mulPost, op, osp, w3j, stream);
  }
  k_final<<<g256((long)NQTOT*24), 256, 0, stream>>>(postO, (float*)d_out);
}
